// LlamaDecoderLayerDAT_19920058318903
// MI455X (gfx1250) — compile-verified, matrix operand order checked by reading, not run on hardware
//
#include <hip/hip_runtime.h>

#define B_     2
#define NQ_    1024
#define C_     2048
#define H_     16
#define DH_    128
#define G_     4
#define CD_    512
#define LR_    24
#define HR_    72
#define INTER_ 64
#define K_     576
#define LT_    1600   // NQ_ + K_

typedef __attribute__((ext_vector_type(16))) __bf16 v16bf;
typedef __attribute__((ext_vector_type(8)))  __bf16 v8bf;
typedef __attribute__((ext_vector_type(8)))  float  v8f;

__device__ __forceinline__ v8f zero8() {
  v8f z = {0.f,0.f,0.f,0.f,0.f,0.f,0.f,0.f};
  return z;
}

// 16x32 bf16 fragment slice for one lane: caller passes p already offset by
// (row*ld + kb + half*8); we fetch elements [0..7] and [16..23].
__device__ __forceinline__ v16bf load_frag16(const __bf16* p) {
  v8bf lo = *(const v8bf*)p;
  v8bf hi = *(const v8bf*)(p + 16);
  v16bf a;
#pragma unroll
  for (int i = 0; i < 8; ++i) { a[i] = lo[i]; a[i + 8] = hi[i]; }
  return a;
}

// ---------------------------------------------------------------------------
// fp32 -> bf16 conversion
// ---------------------------------------------------------------------------
__global__ __launch_bounds__(256) void cvt_bf16_kernel(const float* __restrict__ s,
                                                       __bf16* __restrict__ d, long n) {
  for (long i = (long)blockIdx.x * blockDim.x + threadIdx.x; i < n;
       i += (long)gridDim.x * blockDim.x)
    d[i] = (__bf16)s[i];
}

// ---------------------------------------------------------------------------
// bf16 WMMA GEMM:  D[m,n] = sum_k A[m,k] * W[n,k]  (+ bias[n])
// Block = 128 threads (4 waves). Each wave: 16 rows x 64 cols (4 accumulators).
// Software-pipelined K loop: fragments for step i+1 load while step i multiplies.
// grid = (N/64, ceil(M/64), batches); per-batch strides for A and D. ldD == N.
// ---------------------------------------------------------------------------
template <typename OutT>
__global__ __launch_bounds__(128) void gemm_wmma_kernel(
    const __bf16* __restrict__ A, const __bf16* __restrict__ W,
    const float* __restrict__ bias, OutT* __restrict__ D,
    int M, int N, int Kt, long strideA, long strideD) {
  const int wave = threadIdx.x >> 5;
  const int lane = threadIdx.x & 31;
  const int half = lane >> 4;
  const int lc   = lane & 15;

  const __bf16* Ab = A + (size_t)blockIdx.z * strideA;
  OutT*         Db = D + (size_t)blockIdx.z * strideD;

  int mrow = blockIdx.y * 64 + wave * 16 + lc;
  if (mrow >= M) mrow = M - 1;            // M is a multiple of 64 in all uses; safety clamp
  const int nb = blockIdx.x * 64;

  const __bf16* ar = Ab + (size_t)mrow * Kt + half * 8;
  const __bf16* wr0 = W + (size_t)(nb + 0 * 16 + lc) * Kt + half * 8;
  const __bf16* wr1 = W + (size_t)(nb + 1 * 16 + lc) * Kt + half * 8;
  const __bf16* wr2 = W + (size_t)(nb + 2 * 16 + lc) * Kt + half * 8;
  const __bf16* wr3 = W + (size_t)(nb + 3 * 16 + lc) * Kt + half * 8;

  v8f acc[4];
#pragma unroll
  for (int t = 0; t < 4; ++t) acc[t] = zero8();

  // Prologue loads.
  v16bf af = load_frag16(ar);
  v16bf bf0 = load_frag16(wr0);
  v16bf bf1 = load_frag16(wr1);
  v16bf bf2 = load_frag16(wr2);
  v16bf bf3 = load_frag16(wr3);

  for (int kb = 0; kb < Kt; kb += 32) {
    const int nxt = (kb + 32 < Kt) ? (kb + 32) : kb;   // clamped (redundant last reload)
    __builtin_prefetch(ar + kb + 512, 0, 1);           // global_prefetch_b8
    v16bf afn = load_frag16(ar + nxt);
    v16bf bn0 = load_frag16(wr0 + nxt);
    v16bf bn1 = load_frag16(wr1 + nxt);
    v16bf bn2 = load_frag16(wr2 + nxt);
    v16bf bn3 = load_frag16(wr3 + nxt);
    acc[0] = __builtin_amdgcn_wmma_f32_16x16x32_bf16(false, af, false, bf0, (short)0, acc[0], false, false);
    acc[1] = __builtin_amdgcn_wmma_f32_16x16x32_bf16(false, af, false, bf1, (short)0, acc[1], false, false);
    acc[2] = __builtin_amdgcn_wmma_f32_16x16x32_bf16(false, af, false, bf2, (short)0, acc[2], false, false);
    acc[3] = __builtin_amdgcn_wmma_f32_16x16x32_bf16(false, af, false, bf3, (short)0, acc[3], false, false);
    af = afn; bf0 = bn0; bf1 = bn1; bf2 = bn2; bf3 = bn3;
  }

#pragma unroll
  for (int t = 0; t < 4; ++t) {
    const int nc = nb + t * 16 + lc;
    const float bv = bias ? bias[nc] : 0.f;
#pragma unroll
    for (int r = 0; r < 8; ++r) {
      const int row = blockIdx.y * 64 + wave * 16 + r + 8 * half;
      if (row < M) Db[(size_t)row * N + nc] = (OutT)(acc[t][r] + bv);
    }
  }
}

// ---------------------------------------------------------------------------
// V transpose: v_all [b, key, h*DH+d] (bf16) -> Vt [(b*H+h)*DH + d, key] (bf16)
// LDS-tiled 32x32 so both directions stay coalesced.
// grid = (LT/32, DH/32, B*H), block = 256.
// ---------------------------------------------------------------------------
__global__ __launch_bounds__(256) void transpose_v_kernel(const __bf16* __restrict__ V,
                                                          __bf16* __restrict__ Vt) {
  __shared__ __bf16 tile[32][33];
  const int bh = blockIdx.z;
  const int b = bh / H_, h = bh % H_;
  const int k0 = blockIdx.x * 32;
  const int d0 = blockIdx.y * 32;
  const int tx = threadIdx.x & 31;
  const int ty = threadIdx.x >> 5;     // 0..7
#pragma unroll
  for (int i = ty; i < 32; i += 8)
    tile[i][tx] = V[((size_t)b * LT_ + k0 + i) * C_ + h * DH_ + d0 + tx];
  __syncthreads();
#pragma unroll
  for (int i = ty; i < 32; i += 8)
    Vt[((size_t)bh * DH_ + d0 + i) * LT_ + k0 + tx] = tile[tx][i];
}

// ---------------------------------------------------------------------------
// Intention gate: pooled = mean_HW(lr group tokens); gate = sigmoid(pooled @ Wi^T + bi)
// ---------------------------------------------------------------------------
__global__ __launch_bounds__(256) void gate_kernel(const float* __restrict__ hidden,
                                                   const float* __restrict__ piw,
                                                   const float* __restrict__ pib,
                                                   float* __restrict__ gate) {
  const int n = blockIdx.x;
  const int b = n / G_, g = n % G_;
  __shared__ float pooled[CD_];
  const int tid = threadIdx.x;
  for (int c = tid; c < CD_; c += 256) {
    float s = 0.f;
    for (int t = 0; t < K_; ++t)
      s += hidden[((size_t)b * NQ_ + t) * C_ + g * CD_ + c];
    pooled[c] = s * (1.f / (float)K_);
  }
  __syncthreads();
  if (tid < INTER_) {
    float a = pib[tid];
    for (int c = 0; c < CD_; ++c) a += pooled[c] * piw[tid * CD_ + c];
    gate[n * INTER_ + tid] = 1.f / (1.f + expf(-a));
  }
}

// ---------------------------------------------------------------------------
// Offset network: dw-conv3x3 + silu -> LN(512) -> 1x1(512->64)*gate -> LN(64)
// -> 1x1(64->2); pos = tanh(ref + off). One block per (n, y, x) position.
// ---------------------------------------------------------------------------
__global__ __launch_bounds__(256) void offset_kernel(
    const float* __restrict__ hidden, const float* __restrict__ dw,
    const float* __restrict__ ln1w, const float* __restrict__ ln1b,
    const float* __restrict__ lrproj, const float* __restrict__ gate,
    const float* __restrict__ ln2w, const float* __restrict__ ln2b,
    const float* __restrict__ offw, float* __restrict__ pos) {
  const int blk = blockIdx.x;
  const int p = blk % (LR_ * LR_);
  const int n = blk / (LR_ * LR_);
  const int b = n / G_, g = n % G_;
  const int y = p / LR_, x = p % LR_;
  const int tid = threadIdx.x;

  __shared__ float act[CD_];
  __shared__ float red[256];
  __shared__ float red2[256];
  __shared__ float ybuf[INTER_];

  float lsum = 0.f, lsq = 0.f;
  for (int c = tid; c < CD_; c += 256) {
    float s = 0.f;
#pragma unroll
    for (int ky = -1; ky <= 1; ++ky)
#pragma unroll
      for (int kx = -1; kx <= 1; ++kx) {
        const int yy = y + ky, xx = x + kx;
        if (yy >= 0 && yy < LR_ && xx >= 0 && xx < LR_)
          s += hidden[((size_t)b * NQ_ + (yy * LR_ + xx)) * C_ + g * CD_ + c] *
               dw[c * 9 + (ky + 1) * 3 + (kx + 1)];
      }
    const float a = s / (1.f + expf(-s));   // silu
    act[c] = a;
    lsum += a;
    lsq += a * a;
  }
  red[tid] = lsum; red2[tid] = lsq;
  __syncthreads();
  for (int off = 128; off > 0; off >>= 1) {
    if (tid < off) { red[tid] += red[tid + off]; red2[tid] += red2[tid + off]; }
    __syncthreads();
  }
  const float mu = red[0] * (1.f / (float)CD_);
  const float var = red2[0] * (1.f / (float)CD_) - mu * mu;
  const float rstd = rsqrtf(var + 1e-5f);
  for (int c = tid; c < CD_; c += 256)
    act[c] = (act[c] - mu) * rstd * ln1w[c] + ln1b[c];
  __syncthreads();
  if (tid < INTER_) {
    float a = 0.f;
    for (int c = 0; c < CD_; ++c) a += act[c] * lrproj[tid * CD_ + c];
    ybuf[tid] = a * gate[n * INTER_ + tid];
  }
  __syncthreads();
  if (tid == 0) {
    float s1 = 0.f, s2 = 0.f;
    for (int o = 0; o < INTER_; ++o) { s1 += ybuf[o]; s2 += ybuf[o] * ybuf[o]; }
    const float mu2 = s1 * (1.f / (float)INTER_);
    const float var2 = s2 * (1.f / (float)INTER_) - mu2 * mu2;
    const float rs2 = rsqrtf(var2 + 1e-5f);
    float o0 = 0.f, o1 = 0.f;
    for (int o = 0; o < INTER_; ++o) {
      const float xn = (ybuf[o] - mu2) * rs2 * ln2w[o] + ln2b[o];
      o0 += xn * offw[o];
      o1 += xn * offw[INTER_ + o];
    }
    const float refx = ((x + 0.5f) / (float)(LR_ - 1)) * 2.f - 1.f;
    const float refy = ((y + 0.5f) / (float)(LR_ - 1)) * 2.f - 1.f;
    pos[((size_t)n * K_ + p) * 2 + 0] = tanhf(refx + o0);
    pos[((size_t)n * K_ + p) * 2 + 1] = tanhf(refy + o1);
  }
}

// ---------------------------------------------------------------------------
// Bilinear deformable gather: samp[b,k, g*CD+cd] from image_hd grid (bf16 out).
// ---------------------------------------------------------------------------
__global__ __launch_bounds__(256) void sample_kernel(const float* __restrict__ image_hd,
                                                     const float* __restrict__ pos,
                                                     __bf16* __restrict__ samp) {
  const long total = (long)B_ * K_ * C_;
  for (long idx = (long)blockIdx.x * blockDim.x + threadIdx.x; idx < total;
       idx += (long)gridDim.x * blockDim.x) {
    const int c = (int)(idx % C_);
    const long t = idx / C_;
    const int k = (int)(t % K_);
    const int b = (int)(t / K_);
    const int g = c / CD_;
    const int n = b * G_ + g;
    const float px = pos[((long)n * K_ + k) * 2 + 0];
    const float py = pos[((long)n * K_ + k) * 2 + 1];
    const float xs = (px + 1.f) * 0.5f * (float)(HR_ - 1);
    const float ys = (py + 1.f) * 0.5f * (float)(HR_ - 1);
    const float x0f = floorf(xs), y0f = floorf(ys);
    const float wx = xs - x0f, wy = ys - y0f;
    int x0 = min(max((int)x0f, 0), HR_ - 1);
    int x1 = min(x0 + 1, HR_ - 1);
    int y0 = min(max((int)y0f, 0), HR_ - 1);
    int y1 = min(y0 + 1, HR_ - 1);
    const float* im = image_hd + (long)b * HR_ * HR_ * C_ + c;
    const float v00 = im[(long)(y0 * HR_ + x0) * C_];
    const float v01 = im[(long)(y0 * HR_ + x1) * C_];
    const float v10 = im[(long)(y1 * HR_ + x0) * C_];
    const float v11 = im[(long)(y1 * HR_ + x1) * C_];
    const float val = v00 * (1.f - wx) * (1.f - wy) + v01 * wx * (1.f - wy) +
                      v10 * (1.f - wx) * wy + v11 * wx * wy;
    samp[idx] = (__bf16)val;
  }
}

// ---------------------------------------------------------------------------
// Fused flash attention (causal over text keys, HD keys fully visible).
// One wave32 per (b, h, 16-query tile). bf16 WMMA for QK^T and PV with online
// softmax. K fragments are contiguous in k_all; V fragments are contiguous in
// the pre-transposed Vt, so every operand is two 16-B loads.
// ---------------------------------------------------------------------------
__global__ __launch_bounds__(32) void attn_kernel(const __bf16* __restrict__ Q,
                                                  const __bf16* __restrict__ Kall,
                                                  const __bf16* __restrict__ Vt,
                                                  __bf16* __restrict__ O) {
  const float SCALE = 0.08838834764831845f;   // 1/sqrt(128)
  const int lane = threadIdx.x & 31;
  const int half = lane >> 4;
  const int col  = lane & 15;

  int id = blockIdx.x;
  const int QT = NQ_ / 16;
  const int qt = id % QT; id /= QT;
  const int h  = id % H_;
  const int b  = id / H_;

  const size_t qbase = (size_t)b * NQ_ * C_ + (size_t)h * DH_;
  const size_t kbase = (size_t)b * LT_ * C_ + (size_t)h * DH_;
  const size_t vtbase = (size_t)(b * H_ + h) * DH_ * LT_;

  // Preload Q fragments (16 rows x 128 dh = 4 fragments of 16x32).
  v16bf qf[4];
  {
    const __bf16* qr = Q + qbase + (size_t)(qt * 16 + col) * C_ + half * 8;
#pragma unroll
    for (int j = 0; j < 4; ++j) qf[j] = load_frag16(qr + j * 32);
  }

  v8f oacc[8];
#pragma unroll
  for (int t = 0; t < 8; ++t) oacc[t] = zero8();
  float m[8], l[8];
#pragma unroll
  for (int r = 0; r < 8; ++r) { m[r] = -1e30f; l[r] = 0.f; }

  __shared__ __align__(16) __bf16 Plds[16 * 32];

  const int textTiles = (qt * 16 + 15) / 32 + 1;
  const int totalTiles = textTiles + K_ / 32;

  for (int it = 0; it < totalTiles; ++it) {
    const int kb = (it < textTiles) ? it * 32 : NQ_ + (it - textTiles) * 32;

    // Scores for two 16-key subtiles; batch fragment loads before the WMMAs.
    float ps[2][8];
#pragma unroll
    for (int s = 0; s < 2; ++s) {
      const __bf16* kr = Kall + kbase + (size_t)(kb + s * 16 + col) * C_ + half * 8;
      v16bf kf[4];
#pragma unroll
      for (int j = 0; j < 4; ++j) kf[j] = load_frag16(kr + j * 32);
      v8f S = zero8();
#pragma unroll
      for (int j = 0; j < 4; ++j)
        S = __builtin_amdgcn_wmma_f32_16x16x32_bf16(
            false, qf[j], false, kf[j], (short)0, S, false, false);
      const int cg = kb + s * 16 + col;
#pragma unroll
      for (int r = 0; r < 8; ++r) {
        const int row = qt * 16 + r + 8 * half;
        const bool ok = (cg >= NQ_) || (cg <= row);
        ps[s][r] = ok ? S[r] * SCALE : -1e30f;
      }
    }

    // Online softmax update (row reduce across the 16-lane half).
    float fac[8];
#pragma unroll
    for (int r = 0; r < 8; ++r) {
      float v = fmaxf(ps[0][r], ps[1][r]);
      v = fmaxf(v, __shfl_xor(v, 1, 32));
      v = fmaxf(v, __shfl_xor(v, 2, 32));
      v = fmaxf(v, __shfl_xor(v, 4, 32));
      v = fmaxf(v, __shfl_xor(v, 8, 32));
      const float mn = fmaxf(m[r], v);
      fac[r] = __expf(m[r] - mn);
      const float p0 = __expf(ps[0][r] - mn);
      const float p1 = __expf(ps[1][r] - mn);
      ps[0][r] = p0; ps[1][r] = p1;
      float rs = p0 + p1;
      rs += __shfl_xor(rs, 1, 32);
      rs += __shfl_xor(rs, 2, 32);
      rs += __shfl_xor(rs, 4, 32);
      rs += __shfl_xor(rs, 8, 32);
      l[r] = l[r] * fac[r] + rs;
      m[r] = mn;
    }
#pragma unroll
    for (int t = 0; t < 8; ++t)
#pragma unroll
      for (int r = 0; r < 8; ++r) oacc[t][r] *= fac[r];

    // Relayout P (C-layout) -> A-fragment layout through LDS.
    __syncthreads();
#pragma unroll
    for (int s = 0; s < 2; ++s)
#pragma unroll
      for (int r = 0; r < 8; ++r)
        Plds[(r + 8 * half) * 32 + s * 16 + col] = (__bf16)ps[s][r];
    __syncthreads();
    v16bf af = load_frag16((const __bf16*)Plds + col * 32 + half * 8);

    // PV: 8 output tiles of 16x16 over dh; V fragments contiguous via Vt.
#pragma unroll
    for (int tt = 0; tt < 8; tt += 4) {
      v16bf vf[4];
#pragma unroll
      for (int u = 0; u < 4; ++u)
        vf[u] = load_frag16(Vt + vtbase + (size_t)((tt + u) * 16 + col) * LT_ +
                            kb + half * 8);
#pragma unroll
      for (int u = 0; u < 4; ++u)
        oacc[tt + u] = __builtin_amdgcn_wmma_f32_16x16x32_bf16(
            false, af, false, vf[u], (short)0, oacc[tt + u], false, false);
    }
  }

  // Normalize and write attn output (bf16) at [b, q, h*DH + dh].
#pragma unroll
  for (int t = 0; t < 8; ++t)
#pragma unroll
    for (int r = 0; r < 8; ++r) {
      const int row = qt * 16 + r + 8 * half;
      O[((size_t)b * NQ_ + row) * C_ + h * DH_ + t * 16 + col] =
          (__bf16)(oacc[t][r] / l[r]);
    }
}

// ---------------------------------------------------------------------------
// Host-side orchestration
// ---------------------------------------------------------------------------
// workspace layout (bytes, all 256-aligned)
#define OFF_WQ    0L
#define OFF_WK    8388608L
#define OFF_WV    16777216L
#define OFF_WO    25165824L
#define OFF_KHW   33554432L
#define OFF_VHW   41943040L
#define OFF_HID   50331648L
#define OFF_Q     58720256L
#define OFF_KALL  67108864L
#define OFF_VALL  80216064L
#define OFF_SAMP  93323264L
#define OFF_ATTN  98041856L
#define OFF_GATE  106430464L
#define OFF_POS   106432512L
#define OFF_VT    106469376L   // B*H*DH*LT bf16 = 13,107,200 B

extern "C" void kernel_launch(void* const* d_in, const int* in_sizes, int n_in,
                              void* d_out, int out_size, void* d_ws, size_t ws_size,
                              hipStream_t stream) {
  const float* hidden   = (const float*)d_in[0];
  const float* image_hd = (const float*)d_in[1];
  const float* wq       = (const float*)d_in[2];
  const float* wk       = (const float*)d_in[3];
  const float* wv       = (const float*)d_in[4];
  const float* wo       = (const float*)d_in[5];
  const float* conv_dw  = (const float*)d_in[6];
  const float* ln1w     = (const float*)d_in[7];
  const float* ln1b     = (const float*)d_in[8];
  const float* lrproj   = (const float*)d_in[9];
  const float* piw      = (const float*)d_in[10];
  const float* pib      = (const float*)d_in[11];
  const float* ln2w     = (const float*)d_in[12];
  const float* ln2b     = (const float*)d_in[13];
  const float* offw     = (const float*)d_in[14];
  const float* khw      = (const float*)d_in[15];
  const float* khb      = (const float*)d_in[16];
  const float* vhw      = (const float*)d_in[17];
  const float* vhb      = (const float*)d_in[18];

  char* ws = (char*)d_ws;
  __bf16* wq_bf   = (__bf16*)(ws + OFF_WQ);
  __bf16* wk_bf   = (__bf16*)(ws + OFF_WK);
  __bf16* wv_bf   = (__bf16*)(ws + OFF_WV);
  __bf16* wo_bf   = (__bf16*)(ws + OFF_WO);
  __bf16* khw_bf  = (__bf16*)(ws + OFF_KHW);
  __bf16* vhw_bf  = (__bf16*)(ws + OFF_VHW);
  __bf16* hid_bf  = (__bf16*)(ws + OFF_HID);
  __bf16* q_bf    = (__bf16*)(ws + OFF_Q);
  __bf16* kall_bf = (__bf16*)(ws + OFF_KALL);
  __bf16* vall_bf = (__bf16*)(ws + OFF_VALL);
  __bf16* samp_bf = (__bf16*)(ws + OFF_SAMP);
  __bf16* attn_bf = (__bf16*)(ws + OFF_ATTN);
  float*  gate    = (float*)(ws + OFF_GATE);
  float*  pos     = (float*)(ws + OFF_POS);
  __bf16* vt_bf   = (__bf16*)(ws + OFF_VT);

  const long WN = 2048L * 2048L;            // weight elements
  const long HN = (long)B_ * NQ_ * C_;      // hidden elements

  // fp32 -> bf16 conversions
  cvt_bf16_kernel<<<2048, 256, 0, stream>>>(wq, wq_bf, WN);
  cvt_bf16_kernel<<<2048, 256, 0, stream>>>(wk, wk_bf, WN);
  cvt_bf16_kernel<<<2048, 256, 0, stream>>>(wv, wv_bf, WN);
  cvt_bf16_kernel<<<2048, 256, 0, stream>>>(wo, wo_bf, WN);
  cvt_bf16_kernel<<<2048, 256, 0, stream>>>(khw, khw_bf, WN);
  cvt_bf16_kernel<<<2048, 256, 0, stream>>>(vhw, vhw_bf, WN);
  cvt_bf16_kernel<<<2048, 256, 0, stream>>>(hidden, hid_bf, HN);

  const dim3 blk(128);
  // q = hidden @ wq^T   (fold batch into M)
  gemm_wmma_kernel<__bf16><<<dim3(32, 32, 1), blk, 0, stream>>>(
      hid_bf, wq_bf, nullptr, q_bf, 2048, 2048, 2048, 0L, 0L);
  // k_txt / v_txt into rows [0,1024) of the concatenated K/V buffers
  gemm_wmma_kernel<__bf16><<<dim3(32, 16, 2), blk, 0, stream>>>(
      hid_bf, wk_bf, nullptr, kall_bf, 1024, 2048, 2048,
      (long)NQ_ * C_, (long)LT_ * C_);
  gemm_wmma_kernel<__bf16><<<dim3(32, 16, 2), blk, 0, stream>>>(
      hid_bf, wv_bf, nullptr, vall_bf, 1024, 2048, 2048,
      (long)NQ_ * C_, (long)LT_ * C_);

  // offset network + deformable gather
  gate_kernel<<<B_ * G_, 256, 0, stream>>>(hidden, piw, pib, gate);
  offset_kernel<<<B_ * G_ * LR_ * LR_, 256, 0, stream>>>(
      hidden, conv_dw, ln1w, ln1b, lrproj, gate, ln2w, ln2b, offw, pos);
  sample_kernel<<<4608, 256, 0, stream>>>(image_hd, pos, samp_bf);

  // k_hd / v_hd into rows [1024,1600) of the concatenated K/V buffers
  gemm_wmma_kernel<__bf16><<<dim3(32, 9, 2), blk, 0, stream>>>(
      samp_bf, khw_bf, khb, kall_bf + (long)1024 * C_, 576, 2048, 2048,
      (long)K_ * C_, (long)LT_ * C_);
  gemm_wmma_kernel<__bf16><<<dim3(32, 9, 2), blk, 0, stream>>>(
      samp_bf, vhw_bf, vhb, vall_bf + (long)1024 * C_, 576, 2048, 2048,
      (long)K_ * C_, (long)LT_ * C_);

  // transpose V for contiguous PV fragments: Vt[(b*H+h)*DH + d, key]
  transpose_v_kernel<<<dim3(LT_ / 32, DH_ / 32, B_ * H_), 256, 0, stream>>>(
      vall_bf, vt_bf);

  // fused attention: one wave per (b, h, 16-query tile)
  attn_kernel<<<B_ * H_ * (NQ_ / 16), 32, 0, stream>>>(q_bf, kall_bf, vt_bf, attn_bf);

  // final projection -> fp32 output
  gemm_wmma_kernel<float><<<dim3(32, 32, 1), blk, 0, stream>>>(
      attn_bf, wo_bf, nullptr, (float*)d_out, 2048, 2048, 2048, 0L, 0L);
}
